// EncoderDecoder_80848464379962
// MI455X (gfx1250) — matrix operand order repaired and checked by reading, not compiled, not run on hardware
//
#include <hip/hip_runtime.h>

#define Bsz   8192
#define Tlen  96
#define Fdim  6
#define Hdim  128
#define TOUT  7
#define TF    (Tlen*Fdim)      // 576
#define HSTR  136              // padded bf16 row stride (avoids LDS bank conflicts)
#define EPSc  1e-7f

typedef __attribute__((ext_vector_type(16))) __bf16        v16bf;
typedef __attribute__((ext_vector_type(8)))  float         v8f;
typedef __attribute__((ext_vector_type(8)))  unsigned int  v8u;

struct alignas(16) U4 { unsigned x, y, z, w; };

static_assert(sizeof(v16bf) == 32, "v16bf size");

__device__ __forceinline__ unsigned short f2bf(float x) {
  unsigned u = __float_as_uint(x);
  u += 0x7FFFu + ((u >> 16) & 1u);           // round-to-nearest-even
  return (unsigned short)(u >> 16);
}
__device__ __forceinline__ float bf2f(unsigned short h) {
  return __uint_as_float(((unsigned)h) << 16);
}
__device__ __forceinline__ unsigned pack2(float lo, float hi) {
  unsigned ul = __float_as_uint(lo); ul += 0x7FFFu + ((ul >> 16) & 1u);
  unsigned uh = __float_as_uint(hi); uh += 0x7FFFu + ((uh >> 16) & 1u);
  return (uh & 0xFFFF0000u) | (ul >> 16);
}
__device__ __forceinline__ float sigf(float x)  { return 1.0f / (1.0f + __expf(-x)); }
__device__ __forceinline__ float tanh_f(float x){ return 2.0f / (1.0f + __expf(-2.0f * x)) - 1.0f; }

// ---- A fragment (16x32 bf16, M=lane&15, K contiguous 16 per lane half) from LDS ----
__device__ __forceinline__ v16bf load_a_frag(const unsigned short* hb, int mrow, int hihalf, int kc) {
  const unsigned short* p = hb + mrow * HSTR + kc * 32 + hihalf * 16;  // K = kc*32 + {0..15|16..31}
  U4 a0 = *(const U4*)p;                                               // 8 halves
  U4 a1 = *(const U4*)(p + 8);                                         // K += 8
  v8u r; r[0]=a0.x; r[1]=a0.y; r[2]=a0.z; r[3]=a0.w;
         r[4]=a1.x; r[5]=a1.y; r[6]=a1.z; r[7]=a1.w;
  return __builtin_bit_cast(v16bf, r);
}
// ---- B fragment (32x16): lane n holds W[row_n][col..col+15], f32->bf16 convert ----
__device__ __forceinline__ v16bf load_b_frag(const float* __restrict__ W, int row, int col) {
  const float* p = W + row * Hdim + col;
  v8u r;
#pragma unroll
  for (int j = 0; j < 8; j++) r[j] = pack2(p[2*j], p[2*j+1]);
  return __builtin_bit_cast(v16bf, r);
}
__device__ __forceinline__ v16bf load_b_frag_sum(const float* __restrict__ W1,
                                                 const float* __restrict__ W2, int row, int col) {
  const float* p1 = W1 + row * Hdim + col;
  const float* p2 = W2 + row * Hdim + col;
  v8u r;
#pragma unroll
  for (int j = 0; j < 8; j++) r[j] = pack2(p1[2*j] + p2[2*j], p1[2*j+1] + p2[2*j+1]);
  return __builtin_bit_cast(v16bf, r);
}
__device__ __forceinline__ v8f wmma_bf(v16bf a, v16bf b, v8f c) {
  return __builtin_amdgcn_wmma_f32_16x16x32_bf16(false, a, false, b, (short)0, c, false, false);
}

__global__ void __launch_bounds__(256)
encdec_lstm_kernel(const float* __restrict__ in,   const float* __restrict__ mean,
                   const float* __restrict__ stdv,
                   const float* __restrict__ WihE, const float* __restrict__ WhhE,
                   const float* __restrict__ bihE, const float* __restrict__ bhhE,
                   const float* __restrict__ WihD, const float* __restrict__ WhhD,
                   const float* __restrict__ bihD, const float* __restrict__ bhhD,
                   const float* __restrict__ Wd,   const float* __restrict__ bd,
                   float* __restrict__ out)
{
  // x-stage (encoder phase) and decoder h-history (decoder phase) share LDS.
  __shared__ union {
    float          x[16 * TF];                    // 36,864 B
    unsigned short hist[TOUT][16 * HSTR];         // 30,464 B
  } su;
  __shared__ unsigned short h_s[2][16 * HSTR];    //  8,704 B (h ping-pong, bf16)
  __shared__ unsigned short inp_s[16 * HSTR];     //  4,352 B (relu(h_T) for decoder step 0)

  const int tid   = threadIdx.x;
  const int wave  = tid >> 5;
  const int lane  = tid & 31;
  const int hih   = lane >> 4;          // which K-half this lane holds
  const int lan16 = lane & 15;
  const int m0    = blockIdx.x * 16;    // batch tile base
  const int ncol  = wave * 16 + lan16;  // hidden unit owned by this lane (0..127)

  // ---- stage normalized input tile, clear initial h ----
  for (int e = tid; e < 16 * TF; e += 256) {
    int m = e / TF, rem = e - m * TF;
    int f = rem % Fdim;
    float sd = stdv[(m0 + m) * Fdim + f] + EPSc;
    float mn = mean[(m0 + m) * Fdim + f];
    su.x[e] = (in[(size_t)(m0 + m) * TF + rem] - mn) / sd;
  }
  for (int e = tid; e < 16 * HSTR; e += 256) h_s[0][e] = 0;

  // ---- per-lane small weights: W_ih_enc rows (F=6) and fused biases ----
  float wih[4][6], biaE[4], biaD[4];
#pragma unroll
  for (int g = 0; g < 4; g++) {
    int rg = g * Hdim + ncol;
#pragma unroll
    for (int f = 0; f < Fdim; f++) wih[g][f] = WihE[rg * Fdim + f];
    biaE[g] = bihE[rg] + bhhE[rg];
    biaD[g] = bihD[rg] + bhhD[rg];
  }

  // ---- encoder W_hh fragments: bf16, register-resident, reused 96 steps ----
  v16bf bfr[4][4];
#pragma unroll
  for (int g = 0; g < 4; g++)
#pragma unroll
    for (int kc = 0; kc < 4; kc++)
      bfr[g][kc] = load_b_frag(WhhE, g * Hdim + wave * 16 + lan16, kc * 32 + hih * 16);

  float c[8];
#pragma unroll
  for (int r = 0; r < 8; r++) c[r] = 0.0f;

  __syncthreads();

  // ================= encoder: 96 recurrent steps =================
  for (int t = 0; t < Tlen; t++) {
    const unsigned short* hb = h_s[t & 1];
    unsigned short*       hw = h_s[(t + 1) & 1];

    v16bf a[4];
#pragma unroll
    for (int kc = 0; kc < 4; kc++) a[kc] = load_a_frag(hb, lan16, hih, kc);

    v8f acc[4];
#pragma unroll
    for (int g = 0; g < 4; g++) {
      v8f z = {};
#pragma unroll
      for (int kc = 0; kc < 4; kc++) z = wmma_bf(a[kc], bfr[g][kc], z);
      acc[g] = z;
    }

#pragma unroll
    for (int r = 0; r < 8; r++) {
      int m = r + hih * 8;
      const float* xr = &su.x[m * TF + t * Fdim];
      float xi = 0.f, xf = 0.f, xg = 0.f, xo = 0.f;
#pragma unroll
      for (int f = 0; f < Fdim; f++) {
        float xv = xr[f];
        xi = fmaf(wih[0][f], xv, xi); xf = fmaf(wih[1][f], xv, xf);
        xg = fmaf(wih[2][f], xv, xg); xo = fmaf(wih[3][f], xv, xo);
      }
      float gi = acc[0][r] + xi + biaE[0];
      float gf = acc[1][r] + xf + biaE[1];
      float gg = acc[2][r] + xg + biaE[2];
      float go = acc[3][r] + xo + biaE[3];
      float cn = sigf(gf) * c[r] + sigf(gi) * tanh_f(gg);
      c[r] = cn;
      hw[m * HSTR + ncol] = f2bf(sigf(go) * tanh_f(cn));
    }
    __syncthreads();
  }
  // h_T now in h_s[0] (96 even); build ctx = relu(h_T)
  for (int e = tid; e < 16 * HSTR; e += 256) {
    unsigned short v = h_s[0][e];
    inp_s[e] = (v & 0x8000u) ? (unsigned short)0 : v;
  }
  __syncthreads();

  // ================= decoder step 0: ctx@W_ih + h_T@W_hh =================
  {
    v16bf a[4];
    v8f acc[4];
#pragma unroll
    for (int kc = 0; kc < 4; kc++) a[kc] = load_a_frag(inp_s, lan16, hih, kc);
#pragma unroll
    for (int g = 0; g < 4; g++) {
      v8f z = {};
#pragma unroll
      for (int kc = 0; kc < 4; kc++) {
        v16bf b = load_b_frag(WihD, g * Hdim + wave * 16 + lan16, kc * 32 + hih * 16);
        z = wmma_bf(a[kc], b, z);
      }
      acc[g] = z;
    }
#pragma unroll
    for (int kc = 0; kc < 4; kc++) a[kc] = load_a_frag(h_s[0], lan16, hih, kc);
#pragma unroll
    for (int g = 0; g < 4; g++) {
      v8f z = acc[g];
#pragma unroll
      for (int kc = 0; kc < 4; kc++) {
        v16bf b = load_b_frag(WhhD, g * Hdim + wave * 16 + lan16, kc * 32 + hih * 16);
        z = wmma_bf(a[kc], b, z);
      }
      acc[g] = z;
    }
#pragma unroll
    for (int r = 0; r < 8; r++) {
      int m = r + hih * 8;
      float gi = acc[0][r] + biaD[0];
      float gf = acc[1][r] + biaD[1];
      float gg = acc[2][r] + biaD[2];
      float go = acc[3][r] + biaD[3];
      float cn = sigf(gf) * c[r] + sigf(gi) * tanh_f(gg);
      c[r] = cn;
      su.hist[0][m * HSTR + ncol] = f2bf(sigf(go) * tanh_f(cn));
    }
    __syncthreads();
  }

  // steps 1..6: input == hidden  =>  single GEMM with (W_ih + W_hh), hoisted to regs
#pragma unroll
  for (int g = 0; g < 4; g++)
#pragma unroll
    for (int kc = 0; kc < 4; kc++)
      bfr[g][kc] = load_b_frag_sum(WihD, WhhD, g * Hdim + wave * 16 + lan16, kc * 32 + hih * 16);

  for (int t = 1; t < TOUT; t++) {
    v16bf a[4];
#pragma unroll
    for (int kc = 0; kc < 4; kc++) a[kc] = load_a_frag(su.hist[t - 1], lan16, hih, kc);
    v8f acc[4];
#pragma unroll
    for (int g = 0; g < 4; g++) {
      v8f z = {};
#pragma unroll
      for (int kc = 0; kc < 4; kc++) z = wmma_bf(a[kc], bfr[g][kc], z);
      acc[g] = z;
    }
#pragma unroll
    for (int r = 0; r < 8; r++) {
      int m = r + hih * 8;
      float gi = acc[0][r] + biaD[0];
      float gf = acc[1][r] + biaD[1];
      float gg = acc[2][r] + biaD[2];
      float go = acc[3][r] + biaD[3];
      float cn = sigf(gf) * c[r] + sigf(gi) * tanh_f(gg);
      c[r] = cn;
      su.hist[t][m * HSTR + ncol] = f2bf(sigf(go) * tanh_f(cn));
    }
    __syncthreads();
  }

  // ================= dense head + de-normalize =================
  for (int e = tid; e < 16 * TOUT * Fdim; e += 256) {
    int m = e / (TOUT * Fdim);
    int rem = e - m * (TOUT * Fdim);
    int t = rem / Fdim, f = rem - t * Fdim;
    const unsigned short* hp = &su.hist[t][m * HSTR];
    const float* wd = Wd + f * Hdim;
    float acc = bd[f];
#pragma unroll 8
    for (int k = 0; k < Hdim; k++) acc = fmaf(bf2f(hp[k]), wd[k], acc);
    float sd = stdv[(m0 + m) * Fdim + f] + EPSc;
    float mn = mean[(m0 + m) * Fdim + f];
    out[(size_t)(m0 + m) * (TOUT * Fdim) + rem] = fmaxf(acc, 0.0f) * sd + mn;
  }
}

extern "C" void kernel_launch(void* const* d_in, const int* in_sizes, int n_in,
                              void* d_out, int out_size, void* d_ws, size_t ws_size,
                              hipStream_t stream) {
  (void)in_sizes; (void)n_in; (void)out_size; (void)d_ws; (void)ws_size;
  encdec_lstm_kernel<<<Bsz / 16, 256, 0, stream>>>(
      (const float*)d_in[0],  (const float*)d_in[1],  (const float*)d_in[2],
      (const float*)d_in[3],  (const float*)d_in[4],  (const float*)d_in[5],
      (const float*)d_in[6],  (const float*)d_in[7],  (const float*)d_in[8],
      (const float*)d_in[9],  (const float*)d_in[10], (const float*)d_in[11],
      (const float*)d_in[12], (float*)d_out);
}
